// FullAttention_5317169512546
// MI455X (gfx1250) — matrix operand order repaired and checked by reading, not compiled, not run on hardware
//
#include <hip/hip_runtime.h>

typedef __attribute__((ext_vector_type(16))) _Float16     v16h;
typedef __attribute__((ext_vector_type(8)))  float        v8f;
typedef __attribute__((ext_vector_type(4)))  float        f4;
typedef __attribute__((ext_vector_type(4)))  unsigned int v4u;
typedef __attribute__((ext_vector_type(8)))  int          v8i;
typedef __attribute__((ext_vector_type(4)))  int          v4i;

#define NEG_MASK -987654321.0f

constexpr int B_ = 8, L_ = 1024, S_ = 1024, H_ = 12, E_ = 64, D_ = 64;
constexpr int ROWS  = 32;    // L-rows per workgroup (16 per wave, 2 waves)
constexpr int CHUNK = 64;    // S-rows staged in LDS per iteration
constexpr int SCS   = 1032;  // score-row stride (floats), padded vs 1024
constexpr int STG   = 68;    // staging-row stride (floats); 272B rows keep 16B alignment
constexpr int SMEM_FLOATS = ROWS * SCS + CHUNK * STG;   // 37376 floats = 149504 B

#define HAVE_TDM __has_builtin(__builtin_amdgcn_tensor_load_to_lds)

// Convert two f4's into 8 consecutive f16 lanes of a fragment.
__device__ __forceinline__ void cvt8(v16h& d, int off, f4 a, f4 b) {
  d[off + 0] = (_Float16)a.x; d[off + 1] = (_Float16)a.y;
  d[off + 2] = (_Float16)a.z; d[off + 3] = (_Float16)a.w;
  d[off + 4] = (_Float16)b.x; d[off + 5] = (_Float16)b.y;
  d[off + 6] = (_Float16)b.z; d[off + 7] = (_Float16)b.w;
}

__device__ __forceinline__ void wait_tensor0() {
#if __has_builtin(__builtin_amdgcn_s_wait_tensorcnt)
  __builtin_amdgcn_s_wait_tensorcnt(0);
#else
  asm volatile("s_wait_tensorcnt 0" ::: "memory");
#endif
}

#if HAVE_TDM
// One TDM descriptor: up-to-3D tile, data_size=4B, LDS pad of 4 dwords after
// every 64 dwords (-> 68-float row stride in LDS).
__device__ __forceinline__ void tdm_load_tile(
    uint32_t lds_off, const float* gptr,
    uint32_t tile_d0, uint32_t tile_d1, uint32_t tile_d2,
    uint32_t td0, uint32_t td1, uint32_t td2,
    uint32_t d0_stride, uint32_t d1_stride)
{
  uint64_t ga = (uint64_t)(uintptr_t)gptr;
  v4u g0 = { 1u,                                   // count=1 (valid descriptor)
             lds_off,                              // lds_addr (bytes)
             (uint32_t)ga,                         // global_addr[31:0]
             (uint32_t)(ga >> 32) | (2u << 30) };  // global_addr[56:32] | type=2
  // data_size=4B (code 2) | pad_enable | pad_interval=64 dwords (code 5)
  // | pad_amount=4 dwords (code 3)
  uint32_t w0 = (2u << 16) | (1u << 20) | (5u << 22) | (3u << 25);
  v8i g1 = { (int)w0,
             (int)(td0 << 16),                     // tensor_dim0[15:0] -> bits 63:48
             (int)((td0 >> 16) | (td1 << 16)),     // dim0 hi | dim1 lo
             (int)((td1 >> 16) | (tile_d0 << 16)), // dim1 hi | tile_dim0
             (int)(tile_d1 | (tile_d2 << 16)),     // tile_dim1 | tile_dim2
             (int)d0_stride,                       // tensor_dim0_stride[31:0]
             (int)(d1_stride << 16),               // d0s[47:32]=0 | d1s[15:0]
             (int)(d1_stride >> 16) };             // d1s[47:16]
  v4i g2 = { (int)td2, 0, 0, 0 };                  // tensor_dim2
  v4i g3 = { 0, 0, 0, 0 };
  v8i gx = { 0, 0, 0, 0, 0, 0, 0, 0 };             // uncharacterized 5th operand (clang-23 form)
  __builtin_amdgcn_tensor_load_to_lds(g0, g1, g2, g3, gx, 0);
}
#else
// Fallback: async-DMA row-major copy of a CHUNK x 64 f32 tile into LDS.
__device__ __forceinline__ void stage_k_async(const float* src, uint32_t stg_off, int tid) {
#pragma unroll
  for (int i = 0; i < 16; ++i) {
    int idx4 = tid + i * 64;
    int r = idx4 >> 4, c4 = idx4 & 15;
    asm volatile("global_load_async_to_lds_b128 %0, %1, off"
                 :: "v"(stg_off + (uint32_t)(r * STG + c4 * 4) * 4u),
                    "v"((unsigned long long)(uintptr_t)(src + (size_t)r * H_ * E_ + c4 * 4))
                 : "memory");
  }
  asm volatile("s_wait_asynccnt 0" ::: "memory");
}
// Fallback: manual transposed staging of V chunk (Vt[d][s]).
__device__ __forceinline__ void stage_v_transposed(const float* src, float* stg, int tid) {
#pragma unroll
  for (int i = 0; i < 16; ++i) {
    int idx4 = tid + i * 64;
    int r = idx4 >> 4, c4 = idx4 & 15;             // s = r, d = c4*4..+3
    f4 t = *(const f4*)(src + (size_t)r * H_ * D_ + c4 * 4);
    stg[(c4 * 4 + 0) * STG + r] = t.x;
    stg[(c4 * 4 + 1) * STG + r] = t.y;
    stg[(c4 * 4 + 2) * STG + r] = t.z;
    stg[(c4 * 4 + 3) * STG + r] = t.w;
  }
}
#endif

__global__ __launch_bounds__(64) void lsa_attention_kernel(
    const float* __restrict__ q, const float* __restrict__ k,
    const float* __restrict__ v, const float* __restrict__ scale,
    float* __restrict__ outV, float* __restrict__ outA)
{
  extern __shared__ float smem[];
  float* sc  = smem;              // ROWS x SCS  : scores -> probabilities
  float* stg = smem + ROWS * SCS; // K rows (phase 1) then V^T rows (phase 2)
  const uint32_t stg_off = (uint32_t)(uintptr_t)stg;   // LDS byte offset (low 32 bits)

  const int tid  = threadIdx.x;
  const int lane = tid & 31;
  const int wave = tid >> 5;

  const int blk = blockIdx.x;
  const int lt  = blk & (L_ / ROWS - 1);
  const int h   = (blk / (L_ / ROWS)) % H_;
  const int b   = blk / ((L_ / ROWS) * H_);

  const int rowbase = lt * ROWS + wave * 16;   // first L-row owned by this wave
  const float sh = scale[h];

  const int m16 = lane & 15;   // row (A frag) or column (B/C frags) within tile
  const int hi  = lane >> 4;   // which half of the wave
  const int b8  = hi * 8;      // K-offset base for 16-bit A-matrix layout
  const int b16 = hi * 16;     // K-offset base for contiguous k order; unused (B frags use b8)

  // ---- Q A-fragments: two 16x32 f16 tiles covering E=64 (ISA A-matrix layout) ----
  v16h aq[2];
  {
    const float* qrow = q + (((size_t)b * L_ + (rowbase + m16)) * H_ + h) * E_;
#pragma unroll
    for (int c = 0; c < 2; ++c) {
      const float* p = qrow + c * 32;
      cvt8(aq[c], 0, *(const f4*)(p + b8),      *(const f4*)(p + b8 + 4));
      cvt8(aq[c], 8, *(const f4*)(p + 16 + b8), *(const f4*)(p + 20 + b8));
    }
  }

  // ================= phase 1: scores = scale_h * Q K^T =================
  for (int s0 = 0; s0 < S_; s0 += CHUNK) {
    __syncthreads();
    {
      const float* src = k + (((size_t)b * S_ + s0) * H_ + h) * E_;
      __builtin_prefetch(src + (size_t)CHUNK * H_ * E_ + tid * 16, 0, 1);
#if HAVE_TDM
      if (wave == 0) {   // 2D tile: 64 contiguous floats x 64 rows (stride H*E)
        tdm_load_tile(stg_off, src, 64, 64, 0, 64, 64, 0, H_ * E_, 0);
        wait_tensor0();
      }
#else
      stage_k_async(src, stg_off, tid);
#endif
    }
    __syncthreads();
#pragma unroll
    for (int t = 0; t < 4; ++t) {          // 4 column tiles of 16 keys
      v8f acc = {};
      const float* kr = stg + (t * 16 + m16) * STG;   // key row for this lane's column
#pragma unroll
      for (int c = 0; c < 2; ++c) {        // E = 64 -> two K=32 steps
        v16h bk;
        const float* p = kr + c * 32 + b8;
        cvt8(bk, 0, *(const f4*)(p),      *(const f4*)(p + 4));
        cvt8(bk, 8, *(const f4*)(p + 16), *(const f4*)(p + 20));
        acc = __builtin_amdgcn_wmma_f32_16x16x32_f16(false, aq[c], false, bk,
                                                     (short)0, acc, false, false);
      }
#pragma unroll
      for (int j = 0; j < 8; ++j)          // D-matrix: m = j + 8*hi, n = m16
        sc[(wave * 16 + j + b8) * SCS + s0 + t * 16 + m16] = sh * acc[j];
    }
  }

  __syncthreads();
  // -------- diagonal self-mask (LSA) --------
  if (lane < 16)
    sc[(wave * 16 + lane) * SCS + (rowbase + lane)] = NEG_MASK;
  __syncthreads();

  // -------- two-pass softmax over S; lanes (r, r+16) share row r --------
  float* rowp = sc + (wave * 16 + m16) * SCS + hi * 512;
  f4* r4 = (f4*)rowp;
  float mx = -3.0e38f;
  for (int i = 0; i < 128; ++i) {
    f4 t = r4[i];
    mx = fmaxf(mx, fmaxf(fmaxf(t.x, t.y), fmaxf(t.z, t.w)));
  }
  mx = fmaxf(mx, __shfl_xor(mx, 16, 32));
  float ssum = 0.f;
  for (int i = 0; i < 128; ++i) {
    f4 t = r4[i];
    t.x = __expf(t.x - mx); t.y = __expf(t.y - mx);
    t.z = __expf(t.z - mx); t.w = __expf(t.w - mx);
    r4[i] = t;
    ssum += t.x + t.y + t.z + t.w;
  }
  ssum += __shfl_xor(ssum, 16, 32);
  const float inv = 1.0f / ssum;

  // -------- normalize probabilities in place in LDS --------
  for (int r2 = 0; r2 < 16; ++r2) {
    float invr = __shfl(inv, r2, 32);
    f4* lp = (f4*)(sc + (wave * 16 + r2) * SCS);
#pragma unroll 2
    for (int it = 0; it < 8; ++it) {
      int q4 = it * 32 + lane;
      lp[q4] = lp[q4] * invr;
    }
  }
  asm volatile("s_wait_dscnt 0" ::: "memory");  // LDS writes done before DMA reads them

  // -------- stream A out of LDS via async DMA, non-temporal (403 MB stream) --------
  for (int r2 = 0; r2 < 16; ++r2) {
    const uint32_t lro = (uint32_t)(uintptr_t)(sc + (wave * 16 + r2) * SCS);
    float* gp = outA + (((size_t)b * H_ + h) * L_ + (rowbase + r2)) * S_;
#pragma unroll 2
    for (int it = 0; it < 8; ++it) {
      int scol = it * 128 + lane * 4;
      asm volatile("global_store_async_from_lds_b128 %0, %1, off th:TH_STORE_NT"
                   :: "v"((unsigned long long)(uintptr_t)(gp + scol)),
                      "v"(lro + (uint32_t)scol * 4u)
                   : "memory");
    }
  }
  // drained by later async/tensor waits; sc is read-only from here on

  // ================= phase 2: out = P V =================
  v8f acc2[4] = {};
  for (int s0 = 0; s0 < S_; s0 += CHUNK) {
    __syncthreads();
    {
      const float* src = v + (((size_t)b * S_ + s0) * H_ + h) * D_;
      __builtin_prefetch(src + (size_t)CHUNK * H_ * D_ + tid * 16, 0, 1);
#if HAVE_TDM
      if (wave == 0) {
        // Transposing descriptor: X=1 elem, Y=64 s-steps (stride H*D),
        // Z=64 d-steps (stride 1) -> LDS receives V^T rows (stride 68).
        tdm_load_tile(stg_off, src, 1, 64, 64, 64, 64, 64, H_ * D_, 1);
        wait_tensor0();
      }
#else
      stage_v_transposed(src, stg, tid);
#endif
    }
    __syncthreads();
#pragma unroll
    for (int ss = 0; ss < 2; ++ss) {       // two K=32 steps per 64-row chunk
      v16h ap;                              // A-frag from normalized probabilities
      const float* p = sc + (wave * 16 + m16) * SCS + s0 + ss * 32;
      cvt8(ap, 0, *(const f4*)(p + b8),      *(const f4*)(p + b8 + 4));
      cvt8(ap, 8, *(const f4*)(p + 16 + b8), *(const f4*)(p + 20 + b8));
#pragma unroll
      for (int dt = 0; dt < 4; ++dt) {     // D = 64 -> 4 column tiles
        v16h bv;  // b[e] = Vt[d][s], s = s0+ss*32 + (e<8 ? b8+e : 16+b8+e-8)
        const float* vp = stg + (dt * 16 + m16) * STG + ss * 32 + b8;
        cvt8(bv, 0, *(const f4*)(vp),      *(const f4*)(vp + 4));
        cvt8(bv, 8, *(const f4*)(vp + 16), *(const f4*)(vp + 20));
        acc2[dt] = __builtin_amdgcn_wmma_f32_16x16x32_f16(false, ap, false, bv,
                                                          (short)0, acc2[dt], false, false);
      }
    }
  }

  // -------- write V-out [B,L,H,D], non-temporal --------
#pragma unroll
  for (int dt = 0; dt < 4; ++dt)
#pragma unroll
    for (int j = 0; j < 8; ++j)
      __builtin_nontemporal_store(
          acc2[dt][j],
          outV + (((size_t)b * L_ + (rowbase + j + b8)) * H_ + h) * D_ + dt * 16 + m16);
}

extern "C" void kernel_launch(void* const* d_in, const int* in_sizes, int n_in,
                              void* d_out, int out_size, void* d_ws, size_t ws_size,
                              hipStream_t stream) {
  const float* q  = (const float*)d_in[0];
  const float* k  = (const float*)d_in[1];
  const float* v  = (const float*)d_in[2];
  const float* sc = (const float*)d_in[3];
  float* outV = (float*)d_out;
  float* outA = outV + (size_t)B_ * L_ * H_ * D_;   // tuple (V, A) flattened

  const size_t shmem = (size_t)SMEM_FLOATS * sizeof(float);
  (void)hipFuncSetAttribute((const void*)lsa_attention_kernel,
                            hipFuncAttributeMaxDynamicSharedMemorySize, (int)shmem);

  dim3 grid(B_ * H_ * (L_ / ROWS));   // 3072 workgroups
  dim3 block(64);                     // 2 waves (wave32)
  lsa_attention_kernel<<<grid, block, shmem, stream>>>(q, k, v, sc, outV, outA);
}
